// Attention_75591424410146
// MI455X (gfx1250) — matrix operand order repaired and checked by reading, not compiled, not run on hardware
//
#include <hip/hip_runtime.h>
#include <hip/hip_bf16.h>

// ---------------------------------------------------------------------------
// MI455X (gfx1250) implementation of the positional-encoded attention block.
// wave32, WMMA bf16 (V_WMMA_F32_16X16X32_BF16) for every GEMM,
// GLOBAL_LOAD_ASYNC_TO_LDS_B128 (ASYNCcnt) for bf16 tile staging when the
// toolchain exposes the builtin (guarded by __has_builtin).
// B=2, N=L=2048, E=512, H=8, C=64, D=3, qk head dim = 4*C = 256.
// ---------------------------------------------------------------------------

#define BB 2
#define NN 2048
#define LLEN 2048
#define EE 512
#define HH 8
#define DD 3
#define CC 64
#define QKD 256          // 4*C
#define VP 80            // padded V columns: 64 value cols + 3 yp cols + 13 zero
#define ALPHA_Q 10.0f
#define QK_SCALE 0.0625f // (4C)^-0.5 = 1/16
#define LDPAD 34         // LDS row stride for VALU-staged 32-wide K tiles
#define LDA32 32         // LDS row stride for async-staged tiles (16B aligned)

#if defined(__gfx1250__) && __has_builtin(__builtin_amdgcn_global_load_async_to_lds_b128)
#define HAVE_ASYNC_LDS 1
#endif

typedef __attribute__((ext_vector_type(16))) __bf16          v16bf;
typedef __attribute__((ext_vector_type(16))) unsigned short  v16u;
typedef __attribute__((ext_vector_type(8)))  float           v8f;

#ifdef HAVE_ASYNC_LDS
// Builtin signature (per hipcc diagnostic): v4i addrspace(1)* src,
// v4i addrspace(3)* dst, imm offset, imm cpol.
typedef int v4i_t __attribute__((vector_size(16)));
typedef __attribute__((address_space(1))) v4i_t gv4i_t;
typedef __attribute__((address_space(3))) v4i_t lv4i_t;
#endif

union bfcast { v16u u; v16bf b; };

// fp32 -> bf16 (round-to-nearest-even), raw bits
static __device__ __forceinline__ unsigned short f2bf(float f) {
  unsigned int u = __float_as_uint(f);
  u += 0x7fffu + ((u >> 16) & 1u);
  return (unsigned short)(u >> 16);
}

static __device__ __forceinline__ v8f wmma_bf16(v16u a, v16u b, v8f c) {
  bfcast A; A.u = a;
  bfcast B; B.u = b;
  // (neg_a, A, neg_b, B, c_mod, C, reuse_a, reuse_b)
  return __builtin_amdgcn_wmma_f32_16x16x32_bf16(false, A.b, false, B.b,
                                                 (short)0, c, false, false);
}

// 16B global -> LDS copy: async (ASYNCcnt) when available, else via VGPRs.
static __device__ __forceinline__ void g2lds_b128(const void* g, void* l) {
#ifdef HAVE_ASYNC_LDS
  __builtin_amdgcn_global_load_async_to_lds_b128((gv4i_t*)g, (lv4i_t*)l, 0, 0);
#else
  *(uint4*)l = *(const uint4*)g;
#endif
}

static __device__ __forceinline__ void g2lds_wait() {
#ifdef HAVE_ASYNC_LDS
  __builtin_amdgcn_s_wait_asynccnt(0);
#endif
}

// A fragment: 16x32 bf16, ISA layout:
// lanes 0-15 row M=lane, K pairs {0,1..6,7} in v0..3, {16..23} in v4..7;
// lanes 16-31 same rows, K+8 / K+24.
static __device__ __forceinline__ v16u load_a_frag(const unsigned short* tile,
                                                   int ld, int mbase, int lane) {
  const unsigned short* p = tile + (size_t)(mbase + (lane & 15)) * ld;
  const int kb = (lane < 16) ? 0 : 8;
  v16u a;
#pragma unroll
  for (int j = 0; j < 8; ++j) {
    const int k = ((j < 4) ? 0 : 16) + kb + 2 * (j & 3);
    a[2 * j]     = p[k];
    a[2 * j + 1] = p[k + 1];
  }
  return a;
}

// B fragment: 32x16 bf16; lane holds one output column N=lane%16,
// lanes 0-15 -> K 0..7,16..23, lanes 16-31 -> K 8..15,24..31 (2 K per VGPR).
// LDS tile is [n][k] with stride ld.
static __device__ __forceinline__ v16u load_b_frag(const unsigned short* tile,
                                                   int ld, int nbase, int lane) {
  const unsigned short* p = tile + (size_t)(nbase + (lane & 15)) * ld;
  const int kb = (lane < 16) ? 0 : 8;
  v16u b;
#pragma unroll
  for (int j = 0; j < 8; ++j) {
    const int k = ((j < 4) ? 0 : 16) + kb + 2 * (j & 3);
    b[2 * j]     = p[k];
    b[2 * j + 1] = p[k + 1];
  }
  return b;
}

// ---------------------------------------------------------------------------
// Weight transpose + cvt: W[Nout][K] fp32 -> Wt[K][Nout] bf16
// ---------------------------------------------------------------------------
__global__ __launch_bounds__(256) void wcvt_kernel(const float* __restrict__ W,
                                                   unsigned short* __restrict__ Wt,
                                                   int Nout, int K) {
  size_t idx = (size_t)blockIdx.x * 256 + threadIdx.x;
  if (idx >= (size_t)Nout * K) return;
  int n = (int)(idx % Nout);
  int k = (int)(idx / Nout);
  Wt[idx] = f2bf(W[(size_t)n * K + k]);
}

// ---------------------------------------------------------------------------
// LayerNorm over E=512 per row, output bf16. 128 threads / row.
// ---------------------------------------------------------------------------
__global__ __launch_bounds__(128) void ln_kernel(const float* __restrict__ x,
                                                 const float* __restrict__ g,
                                                 const float* __restrict__ b,
                                                 unsigned short* __restrict__ out) {
  const int row = blockIdx.x;
  const float* xr = x + (size_t)row * EE;
  const int t = threadIdx.x;
  float v[4];
  float s = 0.f;
#pragma unroll
  for (int i = 0; i < 4; ++i) { v[i] = xr[t + 128 * i]; s += v[i]; }
  __shared__ float red[128];
  red[t] = s; __syncthreads();
  for (int o = 64; o > 0; o >>= 1) { if (t < o) red[t] += red[t + o]; __syncthreads(); }
  const float mean = red[0] * (1.f / EE);
  __syncthreads();
  float s2 = 0.f;
#pragma unroll
  for (int i = 0; i < 4; ++i) { float d = v[i] - mean; s2 += d * d; }
  red[t] = s2; __syncthreads();
  for (int o = 64; o > 0; o >>= 1) { if (t < o) red[t] += red[t + o]; __syncthreads(); }
  const float rstd = rsqrtf(red[0] * (1.f / EE) + 1e-5f);
#pragma unroll
  for (int i = 0; i < 4; ++i) {
    const int e = t + 128 * i;
    out[(size_t)row * EE + e] = f2bf((v[i] - mean) * rstd * g[e] + b[e]);
  }
}

// ---------------------------------------------------------------------------
// Generic GEMM: C[M][Nc] = A[M][K](bf16) @ Bt[K][Nc](bf16) + bias, fp32 out.
// Block: 256 thr = 8 waves; tile 64(M) x 128(N); wave = 16x64 via 4 WMMAs.
// A tile staged with async-to-LDS b128; B tile transposed via VGPRs.
// ---------------------------------------------------------------------------
__global__ __launch_bounds__(256) void gemm_nt_kernel(const unsigned short* __restrict__ A,
                                                      const unsigned short* __restrict__ Bt,
                                                      const float* __restrict__ bias,
                                                      float* __restrict__ Cm,
                                                      int M, int K, int Nc) {
  __shared__ unsigned short As[64 * LDA32];   // async staged, 16B aligned rows
  __shared__ unsigned short Bs[128 * LDPAD];
  const int m0 = blockIdx.x * 64;
  const int n0 = blockIdx.y * 128;
  const int tid = threadIdx.x, lane = tid & 31, wave = tid >> 5;
  const int wm = wave >> 1, wn = wave & 1;
  // per-thread 16B chunk of the 64x32 A tile
  const int ar = tid >> 2, ac = (tid & 3) * 8;
  v8f acc[4] = {};
  for (int k0 = 0; k0 < K; k0 += 32) {
    if (k0 + 32 < K)
      __builtin_prefetch(&Bt[(size_t)(k0 + 32) * Nc + n0], 0, 1);
    g2lds_b128(&A[(size_t)(m0 + ar) * K + k0 + ac], &As[ar * LDA32 + ac]);
#pragma unroll
    for (int i = 0; i < 16; ++i) {          // B: 32x128 = 4096, transpose to [n][k]
      const int idx = tid + 256 * i;
      const int kk = idx >> 7, nn = idx & 127;
      Bs[nn * LDPAD + kk] = Bt[(size_t)(k0 + kk) * Nc + n0 + nn];
    }
    g2lds_wait();
    __syncthreads();
    const v16u a = load_a_frag(As, LDA32, wm * 16, lane);
#pragma unroll
    for (int s = 0; s < 4; ++s) {
      const v16u bf = load_b_frag(Bs, LDPAD, wn * 64 + s * 16, lane);
      acc[s] = wmma_bf16(a, bf, acc[s]);
    }
    __syncthreads();
  }
  const int hi8 = (lane >> 4) << 3;
  const int cc  = lane & 15;
#pragma unroll
  for (int s = 0; s < 4; ++s) {
    const int ncol = n0 + wn * 64 + s * 16 + cc;
    const float bv = bias ? bias[ncol] : 0.f;
#pragma unroll
    for (int r = 0; r < 8; ++r) {
      const int mrow = m0 + wm * 16 + r + hi8;
      Cm[(size_t)mrow * Nc + ncol] = acc[s][r] + bv;
    }
  }
}

// ---------------------------------------------------------------------------
// Q assembly: per (b,n,e) recompute positional trig, write 4 bf16 q elems.
// q layout: [B][H][N][256]  (groups: x1*c1 | x1*s1 | x2*c2 | x2*s2)
// ---------------------------------------------------------------------------
__global__ __launch_bounds__(256) void q_assemble_kernel(
    const float* __restrict__ x1, const float* __restrict__ x2,
    const float* __restrict__ xp,
    const float* __restrict__ pW1, const float* __restrict__ pb1,
    const float* __restrict__ pW2, const float* __restrict__ pb2,
    unsigned short* __restrict__ q) {
  size_t idx = (size_t)blockIdx.x * 256 + threadIdx.x;
  if (idx >= (size_t)BB * NN * EE) return;
  const int e = (int)(idx & (EE - 1));
  const size_t bn = idx >> 9;
  const int n = (int)(bn % NN);
  const int b = (int)(bn / NN);
  const float u0 = ALPHA_Q * xp[bn * 3 + 0];
  const float u1 = ALPHA_Q * xp[bn * 3 + 1];
  const float u2 = ALPHA_Q * xp[bn * 3 + 2];
  const float px1 = u0 * pW1[e * 3 + 0] + u1 * pW1[e * 3 + 1] + u2 * pW1[e * 3 + 2] + pb1[e];
  const float px2 = u0 * pW2[e * 3 + 0] + u1 * pW2[e * 3 + 1] + u2 * pW2[e * 3 + 2] + pb2[e];
  const float c1 = cosf(px1), s1 = sinf(px1);
  const float c2 = cosf(px2), s2 = sinf(px2);
  const float xv1 = x1[idx], xv2 = x2[idx];
  const int h = e >> 6, c = e & 63;
  const size_t base = (((size_t)(b * HH + h)) * NN + n) * QKD + c;
  q[base]       = f2bf(xv1 * c1);
  q[base + 64]  = f2bf(xv1 * s1);
  q[base + 128] = f2bf(xv2 * c2);
  q[base + 192] = f2bf(xv2 * s2);
}

// Key side: no bias in the positional projection; groups y1*c1 | y1*s1 | c2 | s2
__global__ __launch_bounds__(256) void k_assemble_kernel(
    const float* __restrict__ y1, const float* __restrict__ yp,
    const float* __restrict__ pW1, const float* __restrict__ pW2,
    unsigned short* __restrict__ kq) {
  size_t idx = (size_t)blockIdx.x * 256 + threadIdx.x;
  if (idx >= (size_t)BB * LLEN * EE) return;
  const int e = (int)(idx & (EE - 1));
  const size_t bl = idx >> 9;
  const int l = (int)(bl % LLEN);
  const int b = (int)(bl / LLEN);
  const float u0 = ALPHA_Q * yp[bl * 3 + 0];
  const float u1 = ALPHA_Q * yp[bl * 3 + 1];
  const float u2 = ALPHA_Q * yp[bl * 3 + 2];
  const float py1 = u0 * pW1[e * 3 + 0] + u1 * pW1[e * 3 + 1] + u2 * pW1[e * 3 + 2];
  const float py2 = u0 * pW2[e * 3 + 0] + u1 * pW2[e * 3 + 1] + u2 * pW2[e * 3 + 2];
  const float c1 = cosf(py1), s1 = sinf(py1);
  const float c2 = cosf(py2), s2 = sinf(py2);
  const float yv1 = y1[idx];
  const int h = e >> 6, c = e & 63;
  const size_t base = (((size_t)(b * HH + h)) * LLEN + l) * QKD + c;
  kq[base]       = f2bf(yv1 * c1);
  kq[base + 64]  = f2bf(yv1 * s1);
  kq[base + 128] = f2bf(c2);
  kq[base + 192] = f2bf(s2);
}

// ---------------------------------------------------------------------------
// Padded V build: vpad[b][h][l][80]  cols 0..63 = v head slice, 64..66 = yp,
// 67..79 = 0.  One WMMA pass then computes attn@V and attn@yp together.
// ---------------------------------------------------------------------------
__global__ __launch_bounds__(256) void vpad_kernel(const float* __restrict__ vtmp,
                                                   const float* __restrict__ yp,
                                                   unsigned short* __restrict__ vpad) {
  size_t idx = (size_t)blockIdx.x * 256 + threadIdx.x;
  if (idx >= (size_t)BB * HH * LLEN * VP) return;
  const int col = (int)(idx % VP);
  const size_t bhl = idx / VP;
  const int l = (int)(bhl % LLEN);
  const size_t bh = bhl / LLEN;
  const int b = (int)(bh / HH), h = (int)(bh % HH);
  unsigned short o = 0;
  if (col < CC) {
    o = f2bf(vtmp[((size_t)b * LLEN + l) * EE + h * CC + col]);
  } else if (col < CC + DD) {
    o = f2bf(yp[((size_t)b * LLEN + l) * DD + (col - CC)]);
  }
  vpad[idx] = o;
}

// ---------------------------------------------------------------------------
// S = (q . k^T) * scale, written straight into the attn output region (fp32).
// grid: x = N/64, y = L/128, z = B*H.  Both tiles staged async-to-LDS.
// ---------------------------------------------------------------------------
__global__ __launch_bounds__(256) void qk_gemm_kernel(const unsigned short* __restrict__ q,
                                                      const unsigned short* __restrict__ kq,
                                                      float* __restrict__ Sout) {
  __shared__ unsigned short As[64 * LDA32];
  __shared__ unsigned short Bs[128 * LDA32];
  const int bh = blockIdx.z;
  const unsigned short* A  = q  + (size_t)bh * NN * QKD;
  const unsigned short* Bk = kq + (size_t)bh * LLEN * QKD;
  float* S = Sout + (size_t)bh * NN * LLEN;
  const int m0 = blockIdx.x * 64;
  const int n0 = blockIdx.y * 128;
  const int tid = threadIdx.x, lane = tid & 31, wave = tid >> 5;
  const int wm = wave >> 1, wn = wave & 1;
  const int ar = tid >> 2, ac = (tid & 3) * 8;    // 16B chunk coords
  v8f acc[4] = {};
  for (int k0 = 0; k0 < QKD; k0 += 32) {
    // A: 64x32 bf16 (one b128 per thread)
    g2lds_b128(&A[(size_t)(m0 + ar) * QKD + k0 + ac], &As[ar * LDA32 + ac]);
    // B: 128x32 bf16, rows l = n0..n0+127 already [n][k] (two b128 per thread)
#pragma unroll
    for (int i = 0; i < 2; ++i) {
      const int idx = tid + 256 * i;
      const int r = idx >> 2, c8 = (idx & 3) * 8;
      g2lds_b128(&Bk[(size_t)(n0 + r) * QKD + k0 + c8], &Bs[r * LDA32 + c8]);
    }
    g2lds_wait();
    __syncthreads();
    const v16u a = load_a_frag(As, LDA32, wm * 16, lane);
#pragma unroll
    for (int s = 0; s < 4; ++s) {
      const v16u bf = load_b_frag(Bs, LDA32, wn * 64 + s * 16, lane);
      acc[s] = wmma_bf16(a, bf, acc[s]);
    }
    __syncthreads();
  }
  const int hi8 = (lane >> 4) << 3;
  const int cc  = lane & 15;
#pragma unroll
  for (int s = 0; s < 4; ++s) {
    const int col = n0 + wn * 64 + s * 16 + cc;
#pragma unroll
    for (int r = 0; r < 8; ++r) {
      const int row = m0 + wm * 16 + r + hi8;
      S[(size_t)row * LLEN + col] = acc[s][r] * QK_SCALE;
    }
  }
}

// ---------------------------------------------------------------------------
// Row softmax + mask renorm, in place on the attn output region.
// Reference semantics: p=softmax(s); t=p*mask; attn=t/(sum t + 1e-8).
// One 256-thread block per (b,h,n) row of 2048.
// ---------------------------------------------------------------------------
__global__ __launch_bounds__(256) void softmax_kernel(float* __restrict__ attn,
                                                      const float* __restrict__ mask) {
  const int row = blockIdx.x;               // bh*N + n
  const int n  = row % NN;
  const int bh = row / NN;
  const int b  = bh / HH;
  float* r = attn + (size_t)row * LLEN;
  const float* mr = mask + ((size_t)b * NN + n) * LLEN;
  const int t = threadIdx.x;
  float v[8];
  float mx = -3.4e38f;
#pragma unroll
  for (int i = 0; i < 8; ++i) { v[i] = r[t + 256 * i]; mx = fmaxf(mx, v[i]); }
  __shared__ float red[256];
  red[t] = mx; __syncthreads();
  for (int o = 128; o > 0; o >>= 1) { if (t < o) red[t] = fmaxf(red[t], red[t + o]); __syncthreads(); }
  mx = red[0]; __syncthreads();
  float s = 0.f;
#pragma unroll
  for (int i = 0; i < 8; ++i) { v[i] = __expf(v[i] - mx); s += v[i]; }
  red[t] = s; __syncthreads();
  for (int o = 128; o > 0; o >>= 1) { if (t < o) red[t] += red[t + o]; __syncthreads(); }
  const float invZ = 1.f / red[0];
  __syncthreads();
  float ms = 0.f;
#pragma unroll
  for (int i = 0; i < 8; ++i) { v[i] = v[i] * invZ * mr[t + 256 * i]; ms += v[i]; }
  red[t] = ms; __syncthreads();
  for (int o = 128; o > 0; o >>= 1) { if (t < o) red[t] += red[t + o]; __syncthreads(); }
  const float inv = 1.f / (red[0] + 1e-8f);
#pragma unroll
  for (int i = 0; i < 8; ++i) r[t + 256 * i] = v[i] * inv;
}

// ---------------------------------------------------------------------------
// ctx[b][h][n][80] = attn @ vpad  (attn fp32 converted to bf16 on the fly —
// keeps the 268 MB attn tensor single-copy; conversion rides the load).
// grid: x = N/128, y = B*H.  8 waves * (16 rows x 5 col-tiles).
// ---------------------------------------------------------------------------
__global__ __launch_bounds__(256) void av_gemm_kernel(const float* __restrict__ attn,
                                                      const unsigned short* __restrict__ vpad,
                                                      float* __restrict__ ctx) {
  __shared__ unsigned short As[128 * LDPAD];
  __shared__ unsigned short Bs[VP * LDPAD];
  const int bh = blockIdx.y;
  const float* A = attn + (size_t)bh * NN * LLEN;
  const unsigned short* Bv = vpad + (size_t)bh * LLEN * VP;
  float* Cc = ctx + (size_t)bh * NN * VP;
  const int m0 = blockIdx.x * 128;
  const int tid = threadIdx.x, lane = tid & 31, wave = tid >> 5;
  v8f acc[5] = {};
  for (int k0 = 0; k0 < LLEN; k0 += 32) {
    if (k0 + 32 < LLEN)
      __builtin_prefetch(&A[(size_t)m0 * LLEN + k0 + 32], 0, 1);
#pragma unroll
    for (int i = 0; i < 16; ++i) {          // attn: 128x32 fp32 -> bf16 LDS
      const int idx = tid + 256 * i;
      const int r = idx >> 5, c = idx & 31;
      As[r * LDPAD + c] = f2bf(A[(size_t)(m0 + r) * LLEN + k0 + c]);
    }
#pragma unroll
    for (int i = 0; i < 10; ++i) {          // vpad: 32x80 -> [n][k] LDS
      const int idx = tid + 256 * i;
      if (idx < 32 * VP) {
        const int kk = idx / VP, nn = idx % VP;
        Bs[nn * LDPAD + kk] = Bv[(size_t)(k0 + kk) * VP + nn];
      }
    }
    __syncthreads();
    const v16u a = load_a_frag(As, LDPAD, wave * 16, lane);
#pragma unroll
    for (int s = 0; s < 5; ++s) {
      const v16u bf = load_b_frag(Bs, LDPAD, s * 16, lane);
      acc[s] = wmma_bf16(a, bf, acc[s]);
    }
    __syncthreads();
  }
  const int hi8 = (lane >> 4) << 3;
  const int cc  = lane & 15;
#pragma unroll
  for (int s = 0; s < 5; ++s) {
    const int col = s * 16 + cc;
#pragma unroll
    for (int r = 0; r < 8; ++r) {
      const int row = m0 + wave * 16 + r + hi8;
      Cc[(size_t)row * VP + col] = acc[s][r];
    }
  }
}

// ---------------------------------------------------------------------------
// Merge heads -> ctxb[B*N][E] bf16 (input of p1 GEMM)
// ---------------------------------------------------------------------------
__global__ __launch_bounds__(256) void ctx_merge_kernel(const float* __restrict__ ctx,
                                                        unsigned short* __restrict__ ctxb) {
  size_t idx = (size_t)blockIdx.x * 256 + threadIdx.x;
  if (idx >= (size_t)BB * NN * EE) return;
  const int e = (int)(idx & (EE - 1));
  const size_t bn = idx >> 9;
  const int n = (int)(bn % NN);
  const int b = (int)(bn / NN);
  const int h = e >> 6, c = e & 63;
  ctxb[idx] = f2bf(ctx[(((size_t)(b * HH + h)) * NN + n) * VP + c]);
}

// o2[b][n][h*3+d] = ctx[...][64+d] - xp[b][n][d]
__global__ __launch_bounds__(256) void o2_kernel(const float* __restrict__ ctx,
                                                 const float* __restrict__ xp,
                                                 float* __restrict__ o2) {
  size_t idx = (size_t)blockIdx.x * 256 + threadIdx.x;
  if (idx >= (size_t)BB * NN * (HH * DD)) return;
  const int j = (int)(idx % (HH * DD));
  const size_t bn = idx / (HH * DD);
  const int n = (int)(bn % NN);
  const int b = (int)(bn / NN);
  const int h = j / DD, d = j % DD;
  o2[idx] = ctx[(((size_t)(b * HH + h)) * NN + n) * VP + CC + d] - xp[bn * DD + d];
}

// out += o2 @ p2W.T + p2b   (24 MACs per output, VALU)
__global__ __launch_bounds__(256) void p2_add_kernel(const float* __restrict__ o2,
                                                     const float* __restrict__ p2W,
                                                     const float* __restrict__ p2b,
                                                     float* __restrict__ out) {
  size_t idx = (size_t)blockIdx.x * 256 + threadIdx.x;
  if (idx >= (size_t)BB * NN * EE) return;
  const int e = (int)(idx & (EE - 1));
  const size_t bn = idx >> 9;
  float acc = p2b[e];
  const float* o = o2 + bn * (HH * DD);
  const float* w = p2W + (size_t)e * (HH * DD);
#pragma unroll
  for (int j = 0; j < HH * DD; ++j) acc += o[j] * w[j];
  out[idx] += acc;
}

// ---------------------------------------------------------------------------
// Launch
// ---------------------------------------------------------------------------
extern "C" void kernel_launch(void* const* d_in, const int* in_sizes, int n_in,
                              void* d_out, int out_size, void* d_ws, size_t ws_size,
                              hipStream_t stream) {
  (void)in_sizes; (void)n_in; (void)out_size; (void)ws_size;

  const float* x    = (const float*)d_in[0];
  const float* xp   = (const float*)d_in[1];
  const float* y    = (const float*)d_in[2];
  const float* yp   = (const float*)d_in[3];
  const float* mask = (const float*)d_in[4];
  const float* pW1  = (const float*)d_in[5];
  const float* pb1  = (const float*)d_in[6];
  const float* pW2  = (const float*)d_in[7];
  const float* pb2  = (const float*)d_in[8];
  const float* qn_g = (const float*)d_in[9];
  const float* qn_b = (const float*)d_in[10];
  const float* qW1  = (const float*)d_in[11];
  const float* qb1  = (const float*)d_in[12];
  const float* qW2  = (const float*)d_in[13];
  const float* qb2  = (const float*)d_in[14];
  const float* kn_g = (const float*)d_in[15];
  const float* kn_b = (const float*)d_in[16];
  const float* kW1  = (const float*)d_in[17];
  const float* kb1  = (const float*)d_in[18];
  const float* vn_g = (const float*)d_in[19];
  const float* vn_b = (const float*)d_in[20];
  const float* vW   = (const float*)d_in[21];
  const float* vb   = (const float*)d_in[22];
  const float* p1W  = (const float*)d_in[23];
  const float* p1b  = (const float*)d_in[24];
  const float* p2W  = (const float*)d_in[25];
  const float* p2b  = (const float*)d_in[26];

  float* out1 = (float*)d_out;                                     // [B,N,E]
  float* attn = (float*)d_out + (size_t)BB * NN * EE;              // [B,H,N,L]

  // workspace carve-up
  char* ws = (char*)d_ws;
  size_t off = 0;
  auto alloc = [&](size_t bytes) { size_t o = off; off += (bytes + 255) & ~(size_t)255; return o; };
  const size_t ROWS = (size_t)BB * NN;   // 4096
  unsigned short* wt_q1 = (unsigned short*)(ws + alloc((size_t)EE * EE * 2));
  unsigned short* wt_q2 = (unsigned short*)(ws + alloc((size_t)EE * EE * 2));
  unsigned short* wt_k1 = (unsigned short*)(ws + alloc((size_t)EE * EE * 2));
  unsigned short* wt_v  = (unsigned short*)(ws + alloc((size_t)EE * EE * 2));
  unsigned short* wt_p1 = (unsigned short*)(ws + alloc((size_t)EE * EE * 2));
  unsigned short* xn    = (unsigned short*)(ws + alloc(ROWS * EE * 2));
  unsigned short* ynk   = (unsigned short*)(ws + alloc(ROWS * EE * 2));
  unsigned short* ynv   = (unsigned short*)(ws + alloc(ROWS * EE * 2));
  float* x1   = (float*)(ws + alloc(ROWS * EE * 4));
  float* x2   = (float*)(ws + alloc(ROWS * EE * 4));
  float* y1   = (float*)(ws + alloc(ROWS * EE * 4));
  float* vtmp = (float*)(ws + alloc(ROWS * EE * 4));
  unsigned short* qb = (unsigned short*)(ws + alloc((size_t)BB * HH * NN * QKD * 2));
  unsigned short* kb = (unsigned short*)(ws + alloc((size_t)BB * HH * LLEN * QKD * 2));
  unsigned short* vp = (unsigned short*)(ws + alloc((size_t)BB * HH * LLEN * VP * 2));
  float* ctx  = (float*)(ws + alloc((size_t)BB * HH * NN * VP * 4));
  unsigned short* ctxb = (unsigned short*)(ws + alloc(ROWS * EE * 2));
  float* o2   = (float*)(ws + alloc(ROWS * (HH * DD) * 4));

  // 1) weights -> bf16, transposed to [K][Nout]
  const int wblocks = (EE * EE + 255) / 256;
  wcvt_kernel<<<wblocks, 256, 0, stream>>>(qW1, wt_q1, EE, EE);
  wcvt_kernel<<<wblocks, 256, 0, stream>>>(qW2, wt_q2, EE, EE);
  wcvt_kernel<<<wblocks, 256, 0, stream>>>(kW1, wt_k1, EE, EE);
  wcvt_kernel<<<wblocks, 256, 0, stream>>>(vW,  wt_v,  EE, EE);
  wcvt_kernel<<<wblocks, 256, 0, stream>>>(p1W, wt_p1, EE, EE);

  // 2) LayerNorms -> bf16
  ln_kernel<<<(int)ROWS, 128, 0, stream>>>(x, qn_g, qn_b, xn);
  ln_kernel<<<(int)ROWS, 128, 0, stream>>>(y, kn_g, kn_b, ynk);
  ln_kernel<<<(int)ROWS, 128, 0, stream>>>(y, vn_g, vn_b, ynv);

  // 3) projection GEMMs (WMMA bf16)
  dim3 gproj(ROWS / 64, EE / 128);
  gemm_nt_kernel<<<gproj, 256, 0, stream>>>(xn,  wt_q1, qb1, x1,   (int)ROWS, EE, EE);
  gemm_nt_kernel<<<gproj, 256, 0, stream>>>(xn,  wt_q2, qb2, x2,   (int)ROWS, EE, EE);
  gemm_nt_kernel<<<gproj, 256, 0, stream>>>(ynk, wt_k1, kb1, y1,   (int)ROWS, EE, EE);
  gemm_nt_kernel<<<gproj, 256, 0, stream>>>(ynv, wt_v,  vb,  vtmp, (int)ROWS, EE, EE);

  // 4) q/k assembly with on-the-fly positional trig; padded V build
  const int ablocks = (int)((ROWS * EE + 255) / 256);
  q_assemble_kernel<<<ablocks, 256, 0, stream>>>(x1, x2, xp, pW1, pb1, pW2, pb2, qb);
  k_assemble_kernel<<<ablocks, 256, 0, stream>>>(y1, yp, pW1, pW2, kb);
  const int vpb = (int)(((size_t)BB * HH * LLEN * VP + 255) / 256);
  vpad_kernel<<<vpb, 256, 0, stream>>>(vtmp, yp, vp);

  // 5) S = qk^T * scale -> attn region
  dim3 gqk(NN / 64, LLEN / 128, BB * HH);
  qk_gemm_kernel<<<gqk, 256, 0, stream>>>(qb, kb, attn);

  // 6) softmax + mask renorm in place
  softmax_kernel<<<BB * HH * NN, 256, 0, stream>>>(attn, mask);

  // 7) ctx = attn @ vpad (out1-context + attn@yp fused)
  dim3 gav(NN / 128, BB * HH);
  av_gemm_kernel<<<gav, 256, 0, stream>>>(attn, vp, ctx);

  // 8) merge heads, build o2
  ctx_merge_kernel<<<ablocks, 256, 0, stream>>>(ctx, ctxb);
  const int o2b = (int)((ROWS * (HH * DD) + 255) / 256);
  o2_kernel<<<o2b, 256, 0, stream>>>(ctx, xp, o2);

  // 9) out1 = ctxb @ p1W.T + p1b  (direct to d_out)
  gemm_nt_kernel<<<gproj, 256, 0, stream>>>(ctxb, wt_p1, p1b, out1, (int)ROWS, EE, EE);

  // 10) out1 += o2 @ p2W.T + p2b
  p2_add_kernel<<<ablocks, 256, 0, stream>>>(o2, p2W, p2b, out1);
}
